// DispLoss_43061342110168
// MI455X (gfx1250) — matrix operand order repaired and checked by reading, not compiled, not run on hardware
//
#include <hip/hip_runtime.h>
#include <hip/hip_bf16.h>
#include <cstddef>

// ---------------------------------------------------------------------------
// DispLoss for MI455X (gfx1250, wave32, WMMA).
// Dominant cost: 8192x8192x1024 Gram matrix (137 GFLOP dense; halved via
// symmetry to ~68 GFLOP). P_f16 = 16 MB -> resides in the 192 MB L2, so the
// problem is matrix-math bound: f16 WMMA (16 KFLOP/instr) with f32
// accumulation, fused exp + row/col reduction epilogue (never materialize
// the 256 MB logits). Staging uses GLOBAL_LOAD_ASYNC_TO_LDS_B128 (ASYNCcnt)
// with double-buffered LDS when the toolchain exposes the builtin.
// ---------------------------------------------------------------------------

#define N_CLASS 8192
#define FEAT    1024
#define BATCH   1024
#define INV_TEMP 10.0f

#define BM 128
#define BN 128
#define BK 32   // == WMMA K for f16

typedef __attribute__((ext_vector_type(16))) _Float16 v16h;
typedef __attribute__((ext_vector_type(8)))  _Float16 v8h;
typedef __attribute__((ext_vector_type(4)))  _Float16 v4h;
typedef __attribute__((ext_vector_type(8)))  float    v8f;
typedef __attribute__((ext_vector_type(4)))  int      v4i;

#if __has_builtin(__builtin_amdgcn_global_load_async_to_lds_b128)
#define HAVE_ASYNC_LDS 1
#define AS_GLOBAL __attribute__((address_space(1)))
#define AS_LOCAL  __attribute__((address_space(3)))
#else
#define HAVE_ASYNC_LDS 0
#endif

#if __has_builtin(__builtin_amdgcn_s_wait_asynccnt)
#define WAIT_ASYNC(n) __builtin_amdgcn_s_wait_asynccnt(n)
#else
#define WAIT_ASYNC(n) asm volatile("s_wait_asynccnt " #n ::: "memory")
#endif

// ---------------------------------------------------------------------------
// Kernel 1: per-class sequential EMA + renormalize (f32), emit f16 matrix,
// zero the rowsum accumulator. One block (256 thr) per class; each thread
// owns 4 contiguous features. Label test is block-uniform -> barriers legal.
// ---------------------------------------------------------------------------
__global__ void __launch_bounds__(256)
ema_update_kernel(const float* __restrict__ feats,
                  const int*   __restrict__ labels,
                  const float* __restrict__ protos,
                  _Float16*    __restrict__ Pf16,
                  float*       __restrict__ rowsum)
{
    const int c = blockIdx.x;
    const int t = threadIdx.x;

    __shared__ int   lab[BATCH];
    __shared__ float red[256];

#pragma unroll
    for (int it = 0; it < BATCH / 256; ++it)
        lab[t + it * 256] = labels[t + it * 256];

    float4 r = ((const float4*)(protos + (size_t)c * FEAT))[t];
    __syncthreads();

    for (int b = 0; b < BATCH; ++b) {
        if (lab[b] == c) {  // uniform across block
            float4 f = ((const float4*)(feats + (size_t)b * FEAT))[t];
            r.x = r.x * 0.95f + f.x * 0.05f;
            r.y = r.y * 0.95f + f.y * 0.05f;
            r.z = r.z * 0.95f + f.z * 0.05f;
            r.w = r.w * 0.95f + f.w * 0.05f;
            red[t] = r.x * r.x + r.y * r.y + r.z * r.z + r.w * r.w;
            __syncthreads();
#pragma unroll
            for (int s = 128; s > 0; s >>= 1) {
                if (t < s) red[t] += red[t + s];
                __syncthreads();
            }
            float inv = 1.0f / fmaxf(sqrtf(red[0]), 1e-12f);
            r.x *= inv; r.y *= inv; r.z *= inv; r.w *= inv;
            __syncthreads();  // protect red[] before next match
        }
    }

    v4h h;
    h[0] = (_Float16)r.x; h[1] = (_Float16)r.y;
    h[2] = (_Float16)r.z; h[3] = (_Float16)r.w;
    ((v4h*)(Pf16 + (size_t)c * FEAT))[t] = h;

    if (t == 0) rowsum[c] = 0.0f;  // ws is poisoned; init accumulator
}

// ---------------------------------------------------------------------------
// Kernel 2: upper-triangle 128x128 Gram tiles. 8 waves per block in a 2x4
// grid, each wave owns a 64x32 sub-tile (4x2 WMMA accumulators). LDS holds
// row-major 128x32 f16 tiles of P for both operand sides; A/B fragments are
// contiguous 16B chunks per the CDNA5 16-bit WMMA lane layouts.
// ---------------------------------------------------------------------------
__global__ void __launch_bounds__(256)
gram_rowsum_kernel(const _Float16* __restrict__ P,
                   float*          __restrict__ rowsum)
{
    const int bRow = blockIdx.y;
    const int bCol = blockIdx.x;
    if (bCol < bRow) return;  // symmetry: upper triangle only
    const bool diagBlock = (bRow == bCol);

#if HAVE_ASYNC_LDS
    __shared__ __align__(16) _Float16 As[2][BM][BK];  // double-buffered
    __shared__ __align__(16) _Float16 Bs[2][BN][BK];
#else
    __shared__ __align__(16) _Float16 As[1][BM][BK];
    __shared__ __align__(16) _Float16 Bs[1][BN][BK];
#endif
    __shared__ float rs_s[BM];
    __shared__ float cs_s[BN];

    const int t       = threadIdx.x;
    const int lane    = t & 31;
    const int wave    = t >> 5;
    const int lhalf   = lane & 15;
    const int hi      = lane >> 4;      // 0: lanes 0-15, 1: lanes 16-31
    const int waveRow = wave >> 2;      // 0..1  -> 64 rows each
    const int waveCol = wave & 3;       // 0..3  -> 32 cols each

    if (t < BM) { rs_s[t] = 0.0f; cs_s[t] = 0.0f; }

    const int ibase = bRow * BM;
    const int jbase = bCol * BN;

    v8f acc[4][2] = {};

    // fragment builder shared by both staging paths
    auto compute_tile = [&](int buf) {
        // A fragment (16x32 f16): lane<16 holds M=lhalf, K={0..7,16..23};
        // lane>=16 holds K={8..15,24..31} -> two contiguous 16B LDS reads.
        v16h afrag[4];
#pragma unroll
        for (int mt = 0; mt < 4; ++mt) {
            const int arow = waveRow * 64 + mt * 16 + lhalf;
            v8h lo = *(const v8h*)(&As[buf][arow][hi * 8]);
            v8h hh = *(const v8h*)(&As[buf][arow][16 + hi * 8]);
#pragma unroll
            for (int e = 0; e < 8; ++e) { afrag[mt][e] = lo[e]; afrag[mt][8 + e] = hh[e]; }
        }
        // B fragment (32x16 f16): lane holds column N=lhalf = one row of P,
        // K split like A (lanes 0-15: K {0..7,16..23}, lanes 16-31: K {8..15,24..31}).
        v16h bfrag[2];
#pragma unroll
        for (int nt = 0; nt < 2; ++nt) {
            const int brow = waveCol * 32 + nt * 16 + lhalf;
            v8h lo = *(const v8h*)(&Bs[buf][brow][hi * 8]);
            v8h hh = *(const v8h*)(&Bs[buf][brow][16 + hi * 8]);
#pragma unroll
            for (int e = 0; e < 8; ++e) { bfrag[nt][e] = lo[e]; bfrag[nt][8 + e] = hh[e]; }
        }
#pragma unroll
        for (int mt = 0; mt < 4; ++mt)
#pragma unroll
            for (int nt = 0; nt < 2; ++nt)
                acc[mt][nt] = __builtin_amdgcn_wmma_f32_16x16x32_f16(
                    false, afrag[mt], false, bfrag[nt],
                    (short)0, acc[mt][nt], false, false);
    };

#if HAVE_ASYNC_LDS
    // ----- async memory->LDS pipeline (4 async issues per wave per tile) ----
    auto issue_async = [&](int kt, int buf) {
        const int k = kt * BK;
#pragma unroll
        for (int it = 0; it < 2; ++it) {
            int id = t + 256 * it;
            int row = id >> 2, ch = id & 3;
            __builtin_amdgcn_global_load_async_to_lds_b128(
                (AS_GLOBAL v4i*)(P + (size_t)(ibase + row) * FEAT + k + ch * 8),
                (AS_LOCAL  v4i*)(&As[buf][row][ch * 8]), 0, 0);
            __builtin_amdgcn_global_load_async_to_lds_b128(
                (AS_GLOBAL v4i*)(P + (size_t)(jbase + row) * FEAT + k + ch * 8),
                (AS_LOCAL  v4i*)(&Bs[buf][row][ch * 8]), 0, 0);
        }
    };

    issue_async(0, 0);
    for (int kt = 0; kt < FEAT / BK; ++kt) {
        const int cur = kt & 1;
        const bool more = (kt + 1 < FEAT / BK);
        if (more) issue_async(kt + 1, cur ^ 1);  // other buffer: safe past kt-1's barrier
        // async loads complete in order per wave: <=4 outstanding means the
        // current buffer's 4 issues have landed in LDS.
        if (more) { WAIT_ASYNC(4); } else { WAIT_ASYNC(0); }
        __syncthreads();
        compute_tile(cur);
        __syncthreads();  // all reads of `cur` done before kt+1 overwrites it
    }
#else
    // ----- fallback: global->reg->LDS staging, software-pipelined ----------
    int4 ra[2], rb[2];
    auto gload = [&](int kt) {
        const int k = kt * BK;
#pragma unroll
        for (int it = 0; it < 2; ++it) {
            int id = t + 256 * it;
            int row = id >> 2, ch = id & 3;
            ra[it] = *(const int4*)(P + (size_t)(ibase + row) * FEAT + k + ch * 8);
            rb[it] = *(const int4*)(P + (size_t)(jbase + row) * FEAT + k + ch * 8);
        }
    };
    auto lstore = [&]() {
#pragma unroll
        for (int it = 0; it < 2; ++it) {
            int id = t + 256 * it;
            int row = id >> 2, ch = id & 3;
            *(int4*)(&As[0][row][ch * 8]) = ra[it];
            *(int4*)(&Bs[0][row][ch * 8]) = rb[it];
        }
    };
    gload(0);
    lstore();
    __syncthreads();
    for (int kt = 0; kt < FEAT / BK; ++kt) {
        if (kt + 1 < FEAT / BK) gload(kt + 1);  // prefetch overlaps WMMA
        compute_tile(0);
        __syncthreads();
        if (kt + 1 < FEAT / BK) {
            lstore();
            __syncthreads();
        }
    }
#endif

    // --- fused epilogue: exp, skip diagonal, reduce rows (and cols if off-diag)
#pragma unroll
    for (int mt = 0; mt < 4; ++mt) {
#pragma unroll
        for (int nt = 0; nt < 2; ++nt) {
            float csum = 0.0f;
#pragma unroll
            for (int v = 0; v < 8; ++v) {
                const int gi = ibase + waveRow * 64 + mt * 16 + v + hi * 8;
                const int gj = jbase + waveCol * 32 + nt * 16 + lhalf;
                float e = __expf(acc[mt][nt][v] * INV_TEMP);
                if (gi == gj) e = 0.0f;   // exclude diagonal exactly
                csum += e;
                // row sum across the 16 lanes holding this row's columns
                float r = e;
                r += __shfl_xor(r, 1);
                r += __shfl_xor(r, 2);
                r += __shfl_xor(r, 4);
                r += __shfl_xor(r, 8);
                if (lhalf == 0)
                    atomicAdd(&rs_s[waveRow * 64 + mt * 16 + v + hi * 8], r);
            }
            if (!diagBlock) {
                // column sums = row sums of the transposed (symmetric) block
                csum += __shfl_xor(csum, 16);
                if (hi == 0)
                    atomicAdd(&cs_s[waveCol * 32 + nt * 16 + lhalf], csum);
            }
        }
    }
    __syncthreads();
    if (t < BM) {
        atomicAdd(&rowsum[ibase + t], rs_s[t]);
        if (!diagBlock) atomicAdd(&rowsum[jbase + t], cs_s[t]);
    }
}

// ---------------------------------------------------------------------------
// Kernel 3: loss = mean_i log(rowsum[i] / (C-1))   (TEMP/BASE_TEMP == 1)
// ---------------------------------------------------------------------------
__global__ void __launch_bounds__(256)
finalize_kernel(const float* __restrict__ rowsum, float* __restrict__ out)
{
    __shared__ float red[256];
    const int t = threadIdx.x;
    float s = 0.0f;
    for (int i = t; i < N_CLASS; i += 256)
        s += logf(rowsum[i] * (1.0f / (float)(N_CLASS - 1)));
    red[t] = s;
    __syncthreads();
#pragma unroll
    for (int k = 128; k > 0; k >>= 1) {
        if (t < k) red[t] += red[t + k];
        __syncthreads();
    }
    if (t == 0) out[0] = red[0] * (1.0f / (float)N_CLASS);
}

// ---------------------------------------------------------------------------
extern "C" void kernel_launch(void* const* d_in, const int* in_sizes, int n_in,
                              void* d_out, int out_size, void* d_ws, size_t ws_size,
                              hipStream_t stream)
{
    const float* feats  = (const float*)d_in[0];   // [1024,1024] f32
    const int*   labels = (const int*)  d_in[1];   // [1024] i32
    const float* protos = (const float*)d_in[2];   // [8192,1024] f32
    float*       out    = (float*)d_out;           // scalar

    _Float16* Pf16   = (_Float16*)d_ws;                                     // 16 MB
    float*    rowsum = (float*)((char*)d_ws +
                                (size_t)N_CLASS * FEAT * sizeof(_Float16)); // 32 KB

    ema_update_kernel<<<N_CLASS, 256, 0, stream>>>(feats, labels, protos, Pf16, rowsum);

    dim3 grid(N_CLASS / BN, N_CLASS / BM);  // upper-triangle blocks do the work
    gram_rowsum_kernel<<<grid, 256, 0, stream>>>(Pf16, rowsum);

    finalize_kernel<<<1, 256, 0, stream>>>(rowsum, out);
}
